// OrderInvariantKernelLinearAttention_89446988907020
// MI455X (gfx1250) — matrix operand order repaired and checked by reading, not compiled, not run on hardware
//
#include <hip/hip_runtime.h>

typedef __attribute__((ext_vector_type(16))) _Float16 v16h;
typedef __attribute__((ext_vector_type(8)))  _Float16 h8;
typedef __attribute__((ext_vector_type(8)))  float    v8f;

#define BH_TOT 32      // B*H
#define HEADS  16
#define LSEQ   4096
#define DDIM   128
#define BS     32
#define NBLK   (LSEQ / BS)   // 128
#define EPS_D  1e-6f

// LDS row strides (in _Float16 units); all *2 bytes are multiples of 16
#define PQ_S 136
#define KT_S 40
#define ST_S 136

#define WS_PER (DDIM * DDIM + DDIM)   // floats per (bh, chunk): G (128x128) + z (128)

union U16 { v16h v; h8 h[2]; };

__device__ __forceinline__ float featmap(float x, float s, float b) {
  float z = s * x + b;
  float y = (z > 0.0f) ? (z + 1.0f) : __expf(z);   // elu(z)+1
  return fminf(fmaxf(y, 0.0f), 10.0f);             // clip [0,10]
}

// ---------------------------------------------------------------------------
// Kernel A: per-chunk local state  G_c = sum phi(K)^T V ,  z_c = colsum phi(K)
// grid = BH_TOT * nchunks, 256 threads (8 waves)
// ---------------------------------------------------------------------------
__launch_bounds__(256, 1)
__global__ void la_chunk_state(const float* __restrict__ K,
                               const float* __restrict__ V,
                               const float* __restrict__ scale,
                               const float* __restrict__ bias,
                               float* __restrict__ ws,
                               int nchunks, int nblk_c)
{
  __shared__ __align__(16) _Float16 sKT[DDIM * KT_S];
  __shared__ __align__(16) _Float16 sVT[DDIM * KT_S];
  __shared__ float sZ[DDIM];

  const int t    = threadIdx.x;
  const int lane = t & 31, wave = t >> 5;
  const int half = lane >> 4, ln16 = lane & 15;
  const int bh   = blockIdx.x / nchunks;
  const int c    = blockIdx.x % nchunks;
  const int h    = bh & (HEADS - 1);

  // k/v mapping: one column, 16 rows per thread (coalesced across threads,
  // transposed LDS store becomes contiguous)
  const int kv_col = t & 127;
  const int kv_rb  = (t >> 7) * 16;
  const float ks = scale[h * DDIM + kv_col];
  const float kb = bias [h * DDIM + kv_col];

  const float* kg = K + (size_t)bh * LSEQ * DDIM;
  const float* vg = V + (size_t)bh * LSEQ * DDIM;

  if (t < DDIM) sZ[t] = 0.0f;
  v8f Sacc[8];
  for (int i = 0; i < 8; ++i) Sacc[i] = {};

  const int n0 = c * nblk_c;
  float rk[16], rv[16];
  {
    size_t off = (size_t)n0 * BS * DDIM;
    for (int j = 0; j < 16; ++j) {
      rk[j] = kg[off + (size_t)(kv_rb + j) * DDIM + kv_col];
      rv[j] = vg[off + (size_t)(kv_rb + j) * DDIM + kv_col];
    }
  }
  __syncthreads();

  for (int i = 0; i < nblk_c; ++i) {
    float pk[16], pv[16];
    for (int j = 0; j < 16; ++j) { pk[j] = featmap(rk[j], ks, kb); pv[j] = rv[j]; }

    if (i + 1 < nblk_c) {   // software-pipelined prefetch of next block
      size_t off = (size_t)(n0 + i + 1) * BS * DDIM;
      for (int j = 0; j < 16; ++j) {
        rk[j] = kg[off + (size_t)(kv_rb + j) * DDIM + kv_col];
        rv[j] = vg[off + (size_t)(kv_rb + j) * DDIM + kv_col];
      }
    }

    __syncthreads();   // B0: previous iteration's LDS reads complete
    {
      h8 a, b2, va, vb2;
      for (int e = 0; e < 8; ++e) {
        a[e]  = (_Float16)pk[e];     b2[e]  = (_Float16)pk[8 + e];
        va[e] = (_Float16)pv[e];     vb2[e] = (_Float16)pv[8 + e];
      }
      *(h8*)&sKT[kv_col * KT_S + kv_rb]     = a;
      *(h8*)&sKT[kv_col * KT_S + kv_rb + 8] = b2;
      *(h8*)&sVT[kv_col * KT_S + kv_rb]     = va;
      *(h8*)&sVT[kv_col * KT_S + kv_rb + 8] = vb2;
      float zp = 0.0f;
      for (int j = 0; j < 16; ++j) zp += pk[j];
      atomicAdd(&sZ[kv_col], zp);
    }
    __syncthreads();   // B1: transposed tiles + Z visible

    {  // S strip update: 8 wmma per wave, accumulators persist in VGPRs
      U16 bv;
      const int d2 = 16 * wave + ln16;
      bv.h[0] = *(const h8*)&sVT[d2 * KT_S + 8 * half];
      bv.h[1] = *(const h8*)&sVT[d2 * KT_S + 16 + 8 * half];
      for (int dt = 0; dt < 8; ++dt) {
        U16 av;
        const int m = dt * 16 + ln16;
        av.h[0] = *(const h8*)&sKT[m * KT_S + 8 * half];
        av.h[1] = *(const h8*)&sKT[m * KT_S + 16 + 8 * half];
        Sacc[dt] = __builtin_amdgcn_wmma_f32_16x16x32_f16(
            false, av.v, false, bv.v, (short)0, Sacc[dt], false, false);
      }
    }
  }

  // Dump state in native accumulator-lane order (layout-agnostic prefix pass)
  float* w = ws + (size_t)blockIdx.x * WS_PER;
  for (int dt = 0; dt < 8; ++dt)
    *(v8f*)&w[(((size_t)wave * 8 + dt) * 32 + lane) * 8] = Sacc[dt];
  if (t < DDIM) w[DDIM * DDIM + t] = sZ[t];
}

// ---------------------------------------------------------------------------
// Kernel B: in-place exclusive prefix over chunks (elementwise). grid = BH_TOT
// ---------------------------------------------------------------------------
__global__ void la_prefix(float* __restrict__ ws, int nchunks)
{
  float* base = ws + (size_t)blockIdx.x * nchunks * WS_PER;
  for (int e = threadIdx.x; e < WS_PER; e += blockDim.x) {
    float run = 0.0f;
    for (int c = 0; c < nchunks; ++c) {
      float* p = base + (size_t)c * WS_PER + e;
      float v = *p; *p = run; run += v;
    }
  }
}

// ---------------------------------------------------------------------------
// Kernel C: seeded block scan producing output. grid = BH_TOT * nchunks
// ---------------------------------------------------------------------------
__launch_bounds__(256, 1)
__global__ void la_scan(const float* __restrict__ Q,
                        const float* __restrict__ K,
                        const float* __restrict__ V,
                        const float* __restrict__ scale,
                        const float* __restrict__ bias,
                        const float* __restrict__ ws,
                        float* __restrict__ Out,
                        int nchunks, int nblk_c)
{
  __shared__ __align__(16) _Float16 sPQ[BS * PQ_S];      //  8704 B
  __shared__ __align__(16) _Float16 sKT[DDIM * KT_S];    // 10240 B
  __shared__ __align__(16) _Float16 sVT[DDIM * KT_S];    // 10240 B
  __shared__ __align__(16) _Float16 sST[DDIM * ST_S];    // 34816 B  (S strip mirror, f16, [d2][d1])
  __shared__ float sZ[DDIM];
  __shared__ float sDen[BS];

  const int t    = threadIdx.x;
  const int lane = t & 31, wave = t >> 5;
  const int half = lane >> 4, ln16 = lane & 15;
  const int bh   = blockIdx.x / nchunks;
  const int c    = blockIdx.x % nchunks;
  const int h    = bh & (HEADS - 1);

  const int q_row = t >> 3;            // 0..31
  const int q_col = (t & 7) * 16;      // 0..112
  const int kv_col = t & 127;
  const int kv_rb  = (t >> 7) * 16;

  float qs[16], qb[16];
  for (int j = 0; j < 16; ++j) {
    qs[j] = scale[h * DDIM + q_col + j];
    qb[j] = bias [h * DDIM + q_col + j];
  }
  const float ks = scale[h * DDIM + kv_col];
  const float kb = bias [h * DDIM + kv_col];

  const float* qg = Q + (size_t)bh * LSEQ * DDIM;
  const float* kg = K + (size_t)bh * LSEQ * DDIM;
  const float* vg = V + (size_t)bh * LSEQ * DDIM;
  float*       og = Out + (size_t)bh * LSEQ * DDIM;

  // Seed state from prefix workspace (zero when single-chunk path)
  v8f Sacc[8];
  if (nchunks > 1) {
    const float* w = ws + (size_t)blockIdx.x * WS_PER;
    for (int dt = 0; dt < 8; ++dt)
      Sacc[dt] = *(const v8f*)&w[(((size_t)wave * 8 + dt) * 32 + lane) * 8];
    if (t < DDIM) sZ[t] = w[DDIM * DDIM + t];
  } else {
    for (int dt = 0; dt < 8; ++dt) Sacc[dt] = {};
    if (t < DDIM) sZ[t] = 0.0f;
  }

  const int n0 = c * nblk_c;
  float rq[16], rk[16], rv[16];
  {
    size_t off = (size_t)n0 * BS * DDIM;
    const float4* qp = (const float4*)(qg + off + (size_t)q_row * DDIM + q_col);
    for (int j4 = 0; j4 < 4; ++j4) {
      float4 x = qp[j4];
      rq[j4 * 4 + 0] = x.x; rq[j4 * 4 + 1] = x.y; rq[j4 * 4 + 2] = x.z; rq[j4 * 4 + 3] = x.w;
    }
    for (int j = 0; j < 16; ++j) {
      rk[j] = kg[off + (size_t)(kv_rb + j) * DDIM + kv_col];
      rv[j] = vg[off + (size_t)(kv_rb + j) * DDIM + kv_col];
    }
  }
  __syncthreads();

  for (int i = 0; i < nblk_c; ++i) {
    float pq[16], pk[16], pv[16];
    for (int j = 0; j < 16; ++j) {
      pq[j] = featmap(rq[j], qs[j], qb[j]);
      pk[j] = featmap(rk[j], ks, kb);
      pv[j] = rv[j];
    }

    if (i + 1 < nblk_c) {  // prefetch next block while this one computes
      size_t off = (size_t)(n0 + i + 1) * BS * DDIM;
      const float4* qp = (const float4*)(qg + off + (size_t)q_row * DDIM + q_col);
      for (int j4 = 0; j4 < 4; ++j4) {
        float4 x = qp[j4];
        rq[j4 * 4 + 0] = x.x; rq[j4 * 4 + 1] = x.y; rq[j4 * 4 + 2] = x.z; rq[j4 * 4 + 3] = x.w;
      }
      for (int j = 0; j < 16; ++j) {
        rk[j] = kg[off + (size_t)(kv_rb + j) * DDIM + kv_col];
        rv[j] = vg[off + (size_t)(kv_rb + j) * DDIM + kv_col];
      }
    }

    __syncthreads();   // B0
    {
      h8 a, b2;
      for (int e = 0; e < 8; ++e) { a[e] = (_Float16)pq[e]; b2[e] = (_Float16)pq[8 + e]; }
      *(h8*)&sPQ[q_row * PQ_S + q_col]     = a;
      *(h8*)&sPQ[q_row * PQ_S + q_col + 8] = b2;
      h8 ka, kb2, va, vb2;
      for (int e = 0; e < 8; ++e) {
        ka[e] = (_Float16)pk[e];   kb2[e] = (_Float16)pk[8 + e];
        va[e] = (_Float16)pv[e];   vb2[e] = (_Float16)pv[8 + e];
      }
      *(h8*)&sKT[kv_col * KT_S + kv_rb]     = ka;
      *(h8*)&sKT[kv_col * KT_S + kv_rb + 8] = kb2;
      *(h8*)&sVT[kv_col * KT_S + kv_rb]     = va;
      *(h8*)&sVT[kv_col * KT_S + kv_rb + 8] = vb2;
      float zp = 0.0f;
      for (int j = 0; j < 16; ++j) zp += pk[j];
      atomicAdd(&sZ[kv_col], zp);
    }
    if (t >= 128 && t < 160) sDen[t - 128] = 0.0f;
    __syncthreads();   // B1

    {  // S += phi_k^T v  (strip per wave)
      U16 bv;
      const int d2 = 16 * wave + ln16;
      bv.h[0] = *(const h8*)&sVT[d2 * KT_S + 8 * half];
      bv.h[1] = *(const h8*)&sVT[d2 * KT_S + 16 + 8 * half];
      for (int dt = 0; dt < 8; ++dt) {
        U16 av;
        const int m = dt * 16 + ln16;
        av.h[0] = *(const h8*)&sKT[m * KT_S + 8 * half];
        av.h[1] = *(const h8*)&sKT[m * KT_S + 16 + 8 * half];
        Sacc[dt] = __builtin_amdgcn_wmma_f32_16x16x32_f16(
            false, av.v, false, bv.v, (short)0, Sacc[dt], false, false);
      }
      // mirror updated strip to LDS as f16, transposed [d2][d1] (wave-local)
      for (int dt = 0; dt < 8; ++dt) {
        h8 o;
        for (int e = 0; e < 8; ++e) o[e] = (_Float16)Sacc[dt][e];
        *(h8*)&sST[d2 * ST_S + dt * 16 + 8 * half] = o;
      }
    }
    {  // den partials: q = t&31, 8 partial-columns of 16
      const int qq = t & 31, part = t >> 5;
      float dp = 0.0f;
      for (int j = 0; j < 16; ++j) {
        int d = part * 16 + j;
        dp += (float)sPQ[qq * PQ_S + d] * sZ[d];
      }
      atomicAdd(&sDen[qq], dp);
    }
    __syncthreads();   // B2

    {  // num = phi_q @ S (strip), then out = num / max(den, eps)
      const int d2 = 16 * wave + ln16;
      v8f acc0 = {}, acc1 = {};
      for (int kt = 0; kt < 4; ++kt) {
        U16 bv, a0, a1;
        bv.h[0] = *(const h8*)&sST[d2 * ST_S + kt * 32 + 8 * half];
        bv.h[1] = *(const h8*)&sST[d2 * ST_S + kt * 32 + 16 + 8 * half];
        a0.h[0] = *(const h8*)&sPQ[ln16 * PQ_S + kt * 32 + 8 * half];
        a0.h[1] = *(const h8*)&sPQ[ln16 * PQ_S + kt * 32 + 16 + 8 * half];
        a1.h[0] = *(const h8*)&sPQ[(16 + ln16) * PQ_S + kt * 32 + 8 * half];
        a1.h[1] = *(const h8*)&sPQ[(16 + ln16) * PQ_S + kt * 32 + 16 + 8 * half];
        acc0 = __builtin_amdgcn_wmma_f32_16x16x32_f16(
            false, a0.v, false, bv.v, (short)0, acc0, false, false);
        acc1 = __builtin_amdgcn_wmma_f32_16x16x32_f16(
            false, a1.v, false, bv.v, (short)0, acc1, false, false);
      }
      float* orow = og + (size_t)(n0 + i) * BS * DDIM;
      for (int r = 0; r < 8; ++r) {
        const int q0 = r + 8 * half;
        const int q1 = 16 + q0;
        const float dn0 = fmaxf(sDen[q0], EPS_D);
        const float dn1 = fmaxf(sDen[q1], EPS_D);
        orow[(size_t)q0 * DDIM + d2] = acc0[r] / dn0;
        orow[(size_t)q1 * DDIM + d2] = acc1[r] / dn1;
      }
    }
  }
}

// ---------------------------------------------------------------------------
extern "C" void kernel_launch(void* const* d_in, const int* in_sizes, int n_in,
                              void* d_out, int out_size, void* d_ws, size_t ws_size,
                              hipStream_t stream) {
  const float* Q  = (const float*)d_in[0];
  const float* K  = (const float*)d_in[1];
  const float* V  = (const float*)d_in[2];
  const float* Sc = (const float*)d_in[3];
  const float* Bi = (const float*)d_in[4];
  float* O  = (float*)d_out;
  float* ws = (float*)d_ws;

  // pick chunk count from available workspace (deterministic in ws_size)
  int C = 1;
  const size_t per_c = (size_t)BH_TOT * WS_PER * sizeof(float);
  const int cands[4] = {16, 8, 4, 2};
  for (int k = 0; k < 4; ++k) {
    if ((size_t)cands[k] * per_c <= ws_size) { C = cands[k]; break; }
  }
  const int nblk_c = NBLK / C;

  if (C > 1) {
    la_chunk_state<<<BH_TOT * C, 256, 0, stream>>>(K, V, Sc, Bi, ws, C, nblk_c);
    la_prefix<<<BH_TOT, 256, 0, stream>>>(ws, C);
  }
  la_scan<<<BH_TOT * C, 256, 0, stream>>>(Q, K, V, Sc, Bi, ws, O, C, nblk_c);
}
